// ALRDLinear_quant_key_v2_23596550324368
// MI455X (gfx1250) — matrix operand order repaired and checked by reading, not compiled, not run on hardware
//
#include <hip/hip_runtime.h>
#include <hip/hip_bf16.h>

// MI455X / gfx1250 implementation.
// y = input @ B_w^T  (bf16 WMMA, f32 accum)  -> ws, with per-tensor amax reduce
// scale = max(amax,1e-8)/3
// out = fakequant(y) @ A_w^T + A_b           (bf16 WMMA, f32 accum)
//
// Workspace layout (requires ~32 MiB + 16 B):
//   [0 .. M*R*4)   : y (fp32, 8192x1024)
//   [M*R*4 .. +4)  : amax bits (uint)
//   [+8  .. +16)   : {scale, inv_scale}
//
// GEMM: 128x128 block tile, 8 wave32s, wave tile 32x64 (2x4 WMMA 16x16x32 bf16),
// double-buffered LDS + register prefetch, one barrier per K-step.

typedef __attribute__((ext_vector_type(16))) __bf16 bf16x16;
typedef __attribute__((ext_vector_type(16))) float  f32x16;
typedef __attribute__((ext_vector_type(8)))  float  f32x8;

static constexpr int PITCH = 40;   // ushorts per LDS row: 32 data + 8 pad (80 B, 16B-aligned)
static constexpr int BM = 128, BN = 128, BK = 32;

__device__ __forceinline__ float fq(float v, float s, float is) {
  // 3-bit symmetric fake quant: clip(round(v/s), -4, 3) * s   (rintf == RNE == jnp.round)
  return fminf(fmaxf(rintf(v * is), -4.0f), 3.0f) * s;
}

union FragU {
  uint4 q[2];
  bf16x16 v;
};

// A fragment 16x32 bf16 (MxK), CDNA5 interleaved half-lane layout:
// lane<16: M=lane, elems 0..7 -> K=0..7, 8..15 -> K=16..23
// lane>=16: M=lane-16, elems 0..7 -> K=8..15, 8..15 -> K=24..31
__device__ __forceinline__ bf16x16 load_frag_a(const unsigned short* tile, int mBase, int lane) {
  const int half = lane >> 4;
  const unsigned short* p = tile + ((lane & 15) + mBase) * PITCH;
  FragU f;
  f.q[0] = *(const uint4*)(p + half * 8);
  f.q[1] = *(const uint4*)(p + 16 + half * 8);
  return f.v;
}

// B fragment 32x16 bf16 (KxN): lane holds column N=lane&15, elems 0..7 -> K=(lane>>4)*8+j, 8..15 -> K=16+(lane>>4)*8+j.
// Tile staged [n][k] so each lane reads two 16 B runs.
__device__ __forceinline__ bf16x16 load_frag_b(const unsigned short* tile, int nBase, int lane) {
  const unsigned short* p = tile + ((lane & 15) + nBase) * PITCH + (lane >> 4) * 8;
  FragU f;
  f.q[0] = *(const uint4*)(p);
  f.q[1] = *(const uint4*)(p + 16);
  return f.v;
}

// 16 fp32 -> 16 bf16 via one vector fptrunc (lowers to in-order v_cvt_pk_bf16_f32 pairs),
// then two 16 B LDS stores. No scalar repacking.
__device__ __forceinline__ void pack_store(unsigned short* dst, const float4* v) {
  union { float4 f[4]; f32x16 w; } s;
  s.f[0] = v[0]; s.f[1] = v[1]; s.f[2] = v[2]; s.f[3] = v[3];
  union { bf16x16 b; uint4 q[2]; } o;
  o.b = __builtin_convertvector(s.w, bf16x16);
  *(uint4*)(dst)     = o.q[0];
  *(uint4*)(dst + 8) = o.q[1];
}

// NT GEMM: C[M,N] = A[M,K] * B[N,K]^T, fp32 in memory, bf16 WMMA, f32 accum.
template <bool QUANT_A, bool HAS_BIAS, bool DO_AMAX>
__global__ __launch_bounds__(256) void gemm_nt_bf16(
    const float* __restrict__ A, const float* __restrict__ B, float* __restrict__ C,
    const float* __restrict__ bias, const float* __restrict__ scales,
    unsigned int* __restrict__ amax, int M, int N, int K) {
  __shared__ __attribute__((aligned(16))) unsigned short As[2][BM * PITCH];
  __shared__ __attribute__((aligned(16))) unsigned short Bs[2][BN * PITCH];

  const int tid  = threadIdx.x;
  const int lane = tid & 31;
  const int wave = tid >> 5;
  const int wm   = wave >> 1;     // 0..3 : wave row (32 rows each)
  const int wn   = wave & 1;      // 0..1 : wave col (64 cols each)
  const int m0   = blockIdx.x * BM;
  const int n0   = blockIdx.y * BN;

  float scale = 1.0f, inv_scale = 1.0f;
  if constexpr (QUANT_A) { scale = scales[0]; inv_scale = scales[1]; }

  // staging: each thread owns 16 consecutive K of one row in BOTH tiles
  const int srow = tid >> 1;            // 0..127
  const int scol = (tid & 1) * 16;      // 0 or 16
  const float* gA = A + (size_t)(m0 + srow) * K + scol;
  const float* gB = B + (size_t)(n0 + srow) * K + scol;

  f32x8 acc[2][4] = {};
  float4 ar[4], br[4];

  // ---- prologue: stage k=0 into buffer 0
#pragma unroll
  for (int q = 0; q < 4; ++q) {
    ar[q] = ((const float4*)gA)[q];
    br[q] = ((const float4*)gB)[q];
  }
  if constexpr (QUANT_A) {
#pragma unroll
    for (int q = 0; q < 4; ++q) {
      ar[q].x = fq(ar[q].x, scale, inv_scale); ar[q].y = fq(ar[q].y, scale, inv_scale);
      ar[q].z = fq(ar[q].z, scale, inv_scale); ar[q].w = fq(ar[q].w, scale, inv_scale);
    }
  }
  pack_store(&As[0][srow * PITCH + scol], ar);
  pack_store(&Bs[0][srow * PITCH + scol], br);
  __syncthreads();

  int buf = 0;
  for (int k0 = 0; k0 < K; k0 += BK) {
    const bool more = (k0 + BK) < K;
    if (more) {  // prefetch next K-slab into registers (uniform branch; EXEC stays full)
      const float* pA = gA + (k0 + BK);
      const float* pB = gB + (k0 + BK);
#pragma unroll
      for (int q = 0; q < 4; ++q) {
        ar[q] = ((const float4*)pA)[q];
        br[q] = ((const float4*)pB)[q];
      }
    }

    // ---- compute on current buffer: 2 A frags, 4 B frags, 8 WMMAs
    const bf16x16 a0 = load_frag_a(&As[buf][0], wm * 32 + 0,  lane);
    const bf16x16 a1 = load_frag_a(&As[buf][0], wm * 32 + 16, lane);
    bf16x16 b[4];
#pragma unroll
    for (int j = 0; j < 4; ++j) b[j] = load_frag_b(&Bs[buf][0], wn * 64 + j * 16, lane);
#pragma unroll
    for (int j = 0; j < 4; ++j) {
      acc[0][j] = __builtin_amdgcn_wmma_f32_16x16x32_bf16(false, a0, false, b[j], (short)0,
                                                          acc[0][j], false, false);
      acc[1][j] = __builtin_amdgcn_wmma_f32_16x16x32_bf16(false, a1, false, b[j], (short)0,
                                                          acc[1][j], false, false);
    }

    if (more) {  // stage prefetched slab into the other buffer
      if constexpr (QUANT_A) {
#pragma unroll
        for (int q = 0; q < 4; ++q) {
          ar[q].x = fq(ar[q].x, scale, inv_scale); ar[q].y = fq(ar[q].y, scale, inv_scale);
          ar[q].z = fq(ar[q].z, scale, inv_scale); ar[q].w = fq(ar[q].w, scale, inv_scale);
        }
      }
      pack_store(&As[buf ^ 1][srow * PITCH + scol], ar);
      pack_store(&Bs[buf ^ 1][srow * PITCH + scol], br);
      __syncthreads();
      buf ^= 1;
    }
  }

  // ---- epilogue. C/D layout: N = lane&15, M = r + (lane>=16 ? 8 : 0)
  const int col0    = n0 + wn * 64 + (lane & 15);
  const int rowBase = m0 + wm * 32 + (lane >> 4) * 8;

  float bj[4] = {0.f, 0.f, 0.f, 0.f};
  if constexpr (HAS_BIAS) {
#pragma unroll
    for (int j = 0; j < 4; ++j) bj[j] = bias[col0 + j * 16];
  }

  float mloc = 0.0f;
#pragma unroll
  for (int i = 0; i < 2; ++i) {
#pragma unroll
    for (int j = 0; j < 4; ++j) {
      const int row = rowBase + i * 16;
      const int col = col0 + j * 16;
#pragma unroll
      for (int r = 0; r < 8; ++r) {
        float v = acc[i][j][r];
        if constexpr (HAS_BIAS) v += bj[j];
        if constexpr (DO_AMAX) mloc = fmaxf(mloc, fabsf(v));
        C[(size_t)(row + r) * N + col] = v;
      }
    }
  }
  if constexpr (DO_AMAX)
    atomicMax(amax, __float_as_uint(mloc));  // values >= 0: uint order == float order
}

__global__ void init_amax_kernel(unsigned int* amax) {
  if (threadIdx.x == 0) *amax = 0u;
}

__global__ void compute_scale_kernel(const unsigned int* __restrict__ amax,
                                     float* __restrict__ scales) {
  const float a = __uint_as_float(*amax);
  const float s = fmaxf(a, 1e-8f) / 3.0f;  // QMAX = 3 for 3-bit symmetric
  scales[0] = s;
  scales[1] = 1.0f / s;
}

extern "C" void kernel_launch(void* const* d_in, const int* in_sizes, int n_in,
                              void* d_out, int out_size, void* d_ws, size_t ws_size,
                              hipStream_t stream) {
  (void)in_sizes; (void)n_in; (void)out_size; (void)ws_size;
  const float* input = (const float*)d_in[0];  // [8192, 4096]
  const float* B_w   = (const float*)d_in[1];  // [1024, 4096]
  const float* A_w   = (const float*)d_in[2];  // [4096, 1024]
  const float* A_b   = (const float*)d_in[3];  // [4096]
  float* out = (float*)d_out;                  // [8192, 4096]

  const int M = 8192, Kin = 4096, R = 1024, Nout = 4096;

  float*        y      = (float*)d_ws;  // M*R fp32
  unsigned int* amax   = (unsigned int*)((char*)d_ws + (size_t)M * R * sizeof(float));
  float*        scales = (float*)((char*)(amax) + 8);

  init_amax_kernel<<<1, 32, 0, stream>>>(amax);

  dim3 blk(256);
  dim3 g1(M / BM, R / BN);
  gemm_nt_bf16<false, false, true><<<g1, blk, 0, stream>>>(
      input, B_w, y, nullptr, nullptr, amax, M, R, Kin);

  compute_scale_kernel<<<1, 1, 0, stream>>>(amax, scales);

  dim3 g2(M / BM, Nout / BN);
  gemm_nt_bf16<true, true, false><<<g2, blk, 0, stream>>>(
      y, A_w, out, A_b, scales, nullptr, M, Nout, R);
}
